// Model_72387378807293
// MI455X (gfx1250) — matrix operand order repaired and checked by reading, not compiled, not run on hardware
//
#include <hip/hip_runtime.h>
#include <hip/hip_bf16.h>

typedef __attribute__((ext_vector_type(16))) _Float16 v16h;
typedef __attribute__((ext_vector_type(8)))  float    v8f;
typedef unsigned int u32x4 __attribute__((ext_vector_type(4)));
typedef int          i32x8 __attribute__((ext_vector_type(8)));
typedef int          i32x4 __attribute__((ext_vector_type(4)));

#define NN 400
#define DD 128
#define KK 16
#define TT 64
#define EE 20000
#define E2 40000
#define NBLK (E2 / 64)   /* 625 edge blocks, 64 edges each */
#define GEPS 1e-10f
#define LEPS 1e-20f

#if __has_builtin(__builtin_amdgcn_tensor_load_to_lds) && __has_builtin(__builtin_amdgcn_s_wait_tensorcnt)
#define USE_TDM 1
#else
#define USE_TDM 0
#endif

__device__ __forceinline__ float sigmoidf_(float x) { return 1.0f / (1.0f + expf(-x)); }
__device__ __forceinline__ float softplusf_(float x) { return (x > 20.0f) ? x : log1pf(expf(x)); }

// ---- WMMA A-fragment loader (CDNA5 ISA §7.12.2, wave32) ----
// 16x32 f16 slice of a row-major [16][128] LDS tile starting at column k0.
// lane L: M=L%16, half=L/16; VGPR j holds K = 16*(j/4) + 8*half + 2*(j%4), +1
__device__ __forceinline__ v16h load_a_frag(const _Float16* sm, int k0, int lane) {
    int m = lane & 15, half = lane >> 4;
    v16h a;
#pragma unroll
    for (int j = 0; j < 8; ++j) {
        int kb = k0 + 16 * (j >> 2) + 8 * half + 2 * (j & 3);
        a[2 * j]     = sm[m * 128 + kb];
        a[2 * j + 1] = sm[m * 128 + kb + 1];
    }
    return a;
}

// ---- TDM: DMA one 16x128 f16 slab of W_dec into LDS (CDNA5 ISA) ----
// 6-arg builtin variant (therock-10.0 headers): (g0, g1, g2, g3, g_extra, cpol)
__device__ __forceinline__ void tdm_load_wdec_tile(const _Float16* gsrc, unsigned lds_off) {
#if USE_TDM
    unsigned long long ga = (unsigned long long)(uintptr_t)gsrc;
    u32x4 g0;
    g0[0] = 1u;                                                   // count=1, user descriptor
    g0[1] = lds_off;                                              // lds_addr (bytes)
    g0[2] = (unsigned)(ga & 0xFFFFFFFFull);                       // global_addr[31:0]
    g0[3] = (unsigned)((ga >> 32) & 0x1FFFFFFull) | (2u << 30);   // global_addr[56:32] | type=2
    i32x8 g1;
    g1[0] = (int)(1u << 16);     // workgroup_mask=0, data_size=1 (2 bytes)
    g1[1] = (int)(128u << 16);   // tensor_dim0 = 128 (elements)
    g1[2] = (int)(400u << 16);   // tensor_dim1 = 400
    g1[3] = (int)(128u << 16);   // tile_dim0 = 128
    g1[4] = 16;                  // tile_dim1 = 16, tile_dim2 = 0
    g1[5] = 128;                 // tensor_dim0_stride = 128
    g1[6] = 0;
    g1[7] = 0;
    i32x4 gz4;
    gz4[0] = 0; gz4[1] = 0; gz4[2] = 0; gz4[3] = 0;               // 2D tensor: groups 2/3 unused
    i32x8 gz8;
    gz8[0] = 0; gz8[1] = 0; gz8[2] = 0; gz8[3] = 0;
    gz8[4] = 0; gz8[5] = 0; gz8[6] = 0; gz8[7] = 0;
    __builtin_amdgcn_tensor_load_to_lds(g0, g1, gz4, gz4, gz8, 0);
#else
    (void)gsrc; (void)lds_off;
#endif
}

// ================= init kernels =================
__global__ __launch_bounds__(128) void init_alpha_kernel(
    const float* __restrict__ mean, const float* __restrict__ stdr,
    const float* __restrict__ eps, float* __restrict__ alpha) {
    int t = threadIdx.x;
    alpha[t] = mean[t] + eps[t] * softplusf_(stdr[t]);
}

__global__ void zero_kernel(float* p, int n) {
    int i = blockIdx.x * blockDim.x + threadIdx.x;
    if (i < n) p[i] = 0.0f;
}

__global__ __launch_bounds__(128) void init_prior_kernel(
    const float* __restrict__ alpha,
    const float* __restrict__ W_sp, const float* __restrict__ b_sp,
    const float* __restrict__ W_sb, const float* __restrict__ b_sb,
    float* __restrict__ phi, float* __restrict__ beta) {
    __shared__ float sa[128];
    int tid = threadIdx.x;
    sa[tid] = alpha[tid];
    __syncthreads();
    int idx = blockIdx.x * 128 + tid;
    if (idx < NN * DD) {
        const float* w = W_sp + (size_t)idx * 128;
        float a = b_sp[idx];
        for (int d = 0; d < 128; ++d) a += sa[d] * w[d];
        phi[idx] = a;
    } else if (idx < NN * DD + KK * DD) {
        int o = idx - NN * DD;
        const float* w = W_sb + (size_t)o * 128;
        float a = b_sb[o];
        for (int d = 0; d < 128; ++d) a += sa[d] * w[d];
        beta[o] = a;
    }
}

__global__ void convert_kernel(const float* __restrict__ W_dec,
                               const float* __restrict__ W_pi,
                               _Float16* __restrict__ wdec16,
                               _Float16* __restrict__ wpi16) {
    int i = blockIdx.x * blockDim.x + threadIdx.x;
    if (i < NN * DD) wdec16[i] = (_Float16)W_dec[i];
    else if (i < NN * DD + KK * DD) wpi16[i - NN * DD] = (_Float16)W_pi[i - NN * DD];
}

// ================= per-step state (GRU + heads) =================
__global__ __launch_bounds__(128) void gru_state_kernel(
    float* phi, float* beta, float* hphi, float* hbeta,
    const float* __restrict__ Wih_n, const float* __restrict__ Whh_n,
    const float* __restrict__ bih_n, const float* __restrict__ bhh_n,
    const float* __restrict__ Wih_c, const float* __restrict__ Whh_c,
    const float* __restrict__ bih_c, const float* __restrict__ bhh_c,
    const float* __restrict__ W_pm, const float* __restrict__ b_pm,
    const float* __restrict__ W_ps, const float* __restrict__ b_ps,
    const float* __restrict__ W_bm, const float* __restrict__ b_bm,
    const float* __restrict__ W_bs, const float* __restrict__ b_bs,
    const float* __restrict__ eps_phi_t, const float* __restrict__ eps_beta_t) {
    __shared__ float sp[128], sh[128], shn[128];
    const int tid = threadIdx.x;
    const int row = blockIdx.x;
    float *state, *h;
    const float *Wih, *Whh, *bih, *bhh, *Wm, *bm, *Ws, *bs, *eps;
    if (row < NN) {
        state = phi + row * DD; h = hphi + row * DD;
        Wih = Wih_n; Whh = Whh_n; bih = bih_n; bhh = bhh_n;
        Wm = W_pm; bm = b_pm; Ws = W_ps; bs = b_ps;
        eps = eps_phi_t + row * DD;
    } else {
        int rb = row - NN;
        state = beta + rb * DD; h = hbeta + rb * DD;
        Wih = Wih_c; Whh = Whh_c; bih = bih_c; bhh = bhh_c;
        Wm = W_bm; bm = b_bm; Ws = W_bs; bs = b_bs;
        eps = eps_beta_t + rb * DD;
    }
    sp[tid] = state[tid];
    sh[tid] = h[tid];
    __syncthreads();
    float gx[3], gh[3];
#pragma unroll
    for (int g = 0; g < 3; ++g) {
        int j = g * 128 + tid;
        const float* wx = Wih + (size_t)j * 256;
        const float* wh = Whh + (size_t)j * 128;
        float ax = bih[j], ah = bhh[j];
        for (int d = 0; d < 128; ++d) {
            ax += sp[d] * (wx[d] + wx[128 + d]);   // x = concat(p, p)
            ah += sh[d] * wh[d];
        }
        gx[g] = ax; gh[g] = ah;
    }
    float rg = sigmoidf_(gx[0] + gh[0]);
    float zg = sigmoidf_(gx[1] + gh[1]);
    float ng = tanhf(gx[2] + rg * gh[2]);
    float hv = (1.0f - zg) * ng + zg * sh[tid];
    h[tid] = hv;
    shn[tid] = hv;
    __syncthreads();
    const float* wm = Wm + (size_t)tid * 128;
    const float* ws = Ws + (size_t)tid * 128;
    float mv = bm[tid], sv = bs[tid];
    for (int d = 0; d < 128; ++d) {
        float hx = shn[d];
        mv += hx * wm[d];
        sv += hx * ws[d];
    }
    state[tid] = mv + eps[tid] * softplusf_(sv);   // sampled phi_s / beta_s
}

// ================= per-step edge kernel =================
// 128 threads = 4 waves; block handles 64 edges (wave w owns 16-edge M-tile w).
// W_dec N-tiles are TDM-staged into LDS (double-buffered) and shared by all 4 waves.
__global__ __launch_bounds__(128) void edge_kernel(
    const int* __restrict__ edges_t,      // [E,2]
    const float* __restrict__ gum_t,      // [2E,16]
    const float* __restrict__ phi,        // [N,D]  phi_s
    const float* __restrict__ beta,       // [K,D]  beta_s
    const _Float16* __restrict__ wdec16,  // [N,D]
    const _Float16* __restrict__ wpi16,   // [K,D]
    const float* __restrict__ b_pi, const float* __restrict__ b_dec,
    float* __restrict__ partial) {
    __shared__ _Float16 s_w[4][2048];     // phi_w tiles (f16)
    __shared__ _Float16 s_pw[4][2048];    // phi_w*phi_c tiles; reused as x after q/prior
    __shared__ float    s_zst[4][256];
    __shared__ _Float16 s_wdec[2][2048];  // TDM-staged W_dec slabs (16 cols x 128 k)
    __shared__ int      s_wi[64], s_ci[64];
    __shared__ float    s_red[4];

    const int tid  = threadIdx.x;
    const int lane = tid & 31;
    const int wv   = tid >> 5;
    const int n    = lane & 15, half = lane >> 4;
    const int blk  = blockIdx.x;

    if (tid < 64) {
        int i = blk * 64 + tid;
        int w, c;
        if (i < EE) { w = edges_t[2 * i];     c = edges_t[2 * i + 1]; }
        else        { int j = i - EE; w = edges_t[2 * j + 1]; c = edges_t[2 * j]; }
        s_wi[tid] = w; s_ci[tid] = c;
    }
    __syncthreads();

    // gather phi_w / phi_c for all 64 edges; pw = phi_w * phi_c
    _Float16* fw  = &s_w[0][0];
    _Float16* fpw = &s_pw[0][0];
    for (int idx = tid; idx < 8192; idx += 128) {
        int m = idx >> 7, d = idx & 127;
        float a = phi[s_wi[m] * 128 + d];
        float b = phi[s_ci[m] * 128 + d];
        fw[idx]  = (_Float16)a;
        fpw[idx] = (_Float16)(a * b);
    }
    __syncthreads();

    const _Float16* mw  = &s_w[wv][0];
    _Float16*       mpw = &s_pw[wv][0];

    // ---- q = pw @ W_pi^T, prior = phi_w @ W_pi^T (4+4 WMMAs per wave) ----
    v8f qacc, pacc;
#pragma unroll
    for (int r = 0; r < 8; ++r) { qacc[r] = 0.0f; pacc[r] = 0.0f; }
#pragma unroll
    for (int kc = 0; kc < 4; ++kc) {
        int k0 = kc * 32;
        v16h aq = load_a_frag(mpw, k0, lane);
        v16h ap = load_a_frag(mw,  k0, lane);
        v16h bf;
#pragma unroll
        for (int j = 0; j < 8; ++j) {
            int k = k0 + 16 * (j >> 2) + 8 * half + 2 * (j & 3);   // B: lane holds col n, K = 16*(j/4) + 8*half + 2*(j%4)
            bf[2 * j]     = wpi16[n * 128 + k];
            bf[2 * j + 1] = wpi16[n * 128 + k + 1];
        }
        qacc = __builtin_amdgcn_wmma_f32_16x16x32_f16(false, aq, false, bf, (short)0, qacc, false, false);
        pacc = __builtin_amdgcn_wmma_f32_16x16x32_f16(false, ap, false, bf, (short)0, pacc, false, false);
    }

    // ---- gumbel-hard softmax + KLD over 16-lane groups ----
    float bpn = b_pi[n];
    float kld = 0.0f;
#pragma unroll
    for (int r = 0; r < 8; ++r) {
        int m = r + 8 * half;
        int eidx = blk * 64 + wv * 16 + m;
        float qv = qacc[r] + bpn;
        float pv = pacc[r] + bpn;
        float u = gum_t[eidx * 16 + n];
        float g = -logf(-logf(u + GEPS) + GEPS);
        float sv = qv + g;                       // TAU == 1
        float mx = sv;
        for (int msk = 1; msk < 16; msk <<= 1) mx = fmaxf(mx, __shfl_xor(mx, msk, 32));
        float ex = expf(sv - mx);
        float sm = ex;
        for (int msk = 1; msk < 16; msk <<= 1) sm += __shfl_xor(sm, msk, 32);
        float y = ex / sm;
        float bv = sv; int bc = n;               // argmax, lowest index wins ties
        for (int msk = 1; msk < 16; msk <<= 1) {
            float ov = __shfl_xor(bv, msk, 32);
            int   oc = __shfl_xor(bc, msk, 32);
            if (ov > bv || (ov == bv && oc < bc)) { bv = ov; bc = oc; }
        }
        float oneh = (n == bc) ? 1.0f : 0.0f;
        s_zst[wv][m * 16 + n] = (oneh - y) + y;  // straight-through value
        float mq = qv;
        for (int msk = 1; msk < 16; msk <<= 1) mq = fmaxf(mq, __shfl_xor(mq, msk, 32));
        float eq = expf(qv - mq);
        float sq = eq;
        for (int msk = 1; msk < 16; msk <<= 1) sq += __shfl_xor(sq, msk, 32);
        float qy = eq / sq;
        float mp = pv;
        for (int msk = 1; msk < 16; msk <<= 1) mp = fmaxf(mp, __shfl_xor(mp, msk, 32));
        float ep = expf(pv - mp);
        float spv = ep;
        for (int msk = 1; msk < 16; msk <<= 1) spv += __shfl_xor(spv, msk, 32);
        float pz = ep / spv;
        kld += qy * (logf(qy + LEPS) - logf(pz + LEPS));
    }
    for (int msk = 1; msk < 32; msk <<= 1) kld += __shfl_xor(kld, msk, 32);

    // ---- x = zst @ beta_s (tiny; f32 accumulate), stored into reused pw buffer ----
    for (int idx = lane; idx < 2048; idx += 32) {
        int m = idx >> 7, d = idx & 127;
        float acc = 0.0f;
#pragma unroll
        for (int k = 0; k < 16; ++k) acc += s_zst[wv][m * 16 + k] * beta[k * 128 + d];
        mpw[idx] = (_Float16)acc;
    }

    // A fragments of x (reused across all 25 N-tiles)
    v16h ax[4];
#pragma unroll
    for (int kc = 0; kc < 4; ++kc) ax[kc] = load_a_frag(mpw, kc * 32, lane);

    // ---- recon = x @ W_dec^T + b_dec with online logsumexp (no recon buffer) ----
    float rmax[8], rsum[8], tgt[8];
#pragma unroll
    for (int r = 0; r < 8; ++r) { rmax[r] = -3.0e38f; rsum[r] = 0.0f; tgt[r] = 0.0f; }

#if USE_TDM
    if (wv == 0) tdm_load_wdec_tile(wdec16, (unsigned)(uintptr_t)&s_wdec[0][0]);
#endif
    for (int nt = 0; nt < 25; ++nt) {
#if USE_TDM
        if (wv == 0) {
            if (nt + 1 < 25) {
                tdm_load_wdec_tile(wdec16 + (size_t)(nt + 1) * 2048,
                                   (unsigned)(uintptr_t)&s_wdec[(nt + 1) & 1][0]);
                __builtin_amdgcn_s_wait_tensorcnt(1);   // slab nt complete
            } else {
                __builtin_amdgcn_s_wait_tensorcnt(0);
            }
        }
#else
        for (int idx = tid; idx < 2048; idx += 128)
            s_wdec[nt & 1][idx] = wdec16[(size_t)nt * 2048 + idx];
#endif
        __syncthreads();
        const _Float16* wd = &s_wdec[nt & 1][0];
        int col = nt * 16 + n;
        v8f acc;
        float bd = b_dec[col];
#pragma unroll
        for (int r = 0; r < 8; ++r) acc[r] = bd;
#pragma unroll
        for (int kc = 0; kc < 4; ++kc) {
            v16h bf;
#pragma unroll
            for (int j = 0; j < 8; ++j) {
                int k = kc * 32 + 16 * (j >> 2) + 8 * half + 2 * (j & 3);
                bf[2 * j]     = wd[n * 128 + k];
                bf[2 * j + 1] = wd[n * 128 + k + 1];
            }
            acc = __builtin_amdgcn_wmma_f32_16x16x32_f16(false, ax[kc], false, bf, (short)0, acc, false, false);
        }
#pragma unroll
        for (int r = 0; r < 8; ++r) {
            float v = acc[r];
            float tmax = v;
            for (int msk = 1; msk < 16; msk <<= 1) tmax = fmaxf(tmax, __shfl_xor(tmax, msk, 32));
            float te = expf(v - tmax);
            float tsum = te;
            for (int msk = 1; msk < 16; msk <<= 1) tsum += __shfl_xor(tsum, msk, 32);
            float nm = fmaxf(rmax[r], tmax);
            rsum[r] = rsum[r] * expf(rmax[r] - nm) + tsum * expf(tmax - nm);
            rmax[r] = nm;
            int cm = s_ci[wv * 16 + r + 8 * half];
            float tv = (col == cm) ? v : 0.0f;
            for (int msk = 1; msk < 16; msk <<= 1) tv += __shfl_xor(tv, msk, 32);
            tgt[r] += tv;
        }
        __syncthreads();
    }

    float bce = 0.0f;
    if (n == 0) {
#pragma unroll
        for (int r = 0; r < 8; ++r) bce += (rmax[r] + logf(rsum[r])) - tgt[r];
    }
    for (int msk = 1; msk < 32; msk <<= 1) bce += __shfl_xor(bce, msk, 32);

    if (lane == 0) s_red[wv] = bce + kld;
    __syncthreads();
    if (tid == 0) partial[blk] += (s_red[0] + s_red[1] + s_red[2] + s_red[3]);
}

__global__ __launch_bounds__(256) void finalize_kernel(const float* __restrict__ partial,
                                                       float* __restrict__ out) {
    __shared__ float red[256];
    int tid = threadIdx.x;
    float s = 0.0f;
    for (int i = tid; i < NBLK; i += 256) s += partial[i];
    red[tid] = s;
    __syncthreads();
    for (int w = 128; w > 0; w >>= 1) {
        if (tid < w) red[tid] += red[tid + w];
        __syncthreads();
    }
    if (tid == 0) out[0] = red[0] * (1.0f / (float)E2);
}

// ================= host launcher =================
extern "C" void kernel_launch(void* const* d_in, const int* in_sizes, int n_in,
                              void* d_out, int out_size, void* d_ws, size_t ws_size,
                              hipStream_t stream) {
    const int*   edges      = (const int*)d_in[0];
    const float* alpha_mean = (const float*)d_in[1];
    const float* alpha_std  = (const float*)d_in[2];
    const float* eps_alpha  = (const float*)d_in[3];
    const float* eps_phi    = (const float*)d_in[4];
    const float* eps_beta   = (const float*)d_in[5];
    const float* gumbel_u   = (const float*)d_in[6];
    const float* W_sp = (const float*)d_in[7];   const float* b_sp = (const float*)d_in[8];
    const float* W_sb = (const float*)d_in[9];   const float* b_sb = (const float*)d_in[10];
    const float* Wih_n = (const float*)d_in[11]; const float* Whh_n = (const float*)d_in[12];
    const float* bih_n = (const float*)d_in[13]; const float* bhh_n = (const float*)d_in[14];
    const float* Wih_c = (const float*)d_in[15]; const float* Whh_c = (const float*)d_in[16];
    const float* bih_c = (const float*)d_in[17]; const float* bhh_c = (const float*)d_in[18];
    const float* W_bm = (const float*)d_in[19];  const float* b_bm = (const float*)d_in[20];
    const float* W_bs = (const float*)d_in[21];  const float* b_bs = (const float*)d_in[22];
    const float* W_pm = (const float*)d_in[23];  const float* b_pm = (const float*)d_in[24];
    const float* W_ps = (const float*)d_in[25];  const float* b_ps = (const float*)d_in[26];
    const float* W_pi = (const float*)d_in[27];  const float* b_pi = (const float*)d_in[28];
    const float* W_dec = (const float*)d_in[29]; const float* b_dec = (const float*)d_in[30];

    // workspace layout
    float* alpha   = (float*)d_ws;            // 128
    float* phi     = alpha + 128;             // 51200 (state buffer, prior -> phi_s)
    float* beta    = phi + NN * DD;           // 2048
    float* hphi    = beta + KK * DD;          // 51200  } contiguous zero region
    float* hbeta   = hphi + NN * DD;          // 2048   }
    float* partial = hbeta + KK * DD;         // NBLK   }
    _Float16* wdec16 = (_Float16*)(partial + NBLK);  // 51200 halves
    _Float16* wpi16  = wdec16 + NN * DD;             // 2048 halves

    const int zero_n = NN * DD + KK * DD + NBLK;

    init_alpha_kernel<<<1, 128, 0, stream>>>(alpha_mean, alpha_std, eps_alpha, alpha);
    zero_kernel<<<(zero_n + 255) / 256, 256, 0, stream>>>(hphi, zero_n);
    init_prior_kernel<<<(NN * DD + KK * DD) / 128, 128, 0, stream>>>(
        alpha, W_sp, b_sp, W_sb, b_sb, phi, beta);
    convert_kernel<<<(NN * DD + KK * DD + 255) / 256, 256, 0, stream>>>(
        W_dec, W_pi, wdec16, wpi16);

    for (int t = 0; t < TT; ++t) {
        gru_state_kernel<<<NN + KK, 128, 0, stream>>>(
            phi, beta, hphi, hbeta,
            Wih_n, Whh_n, bih_n, bhh_n,
            Wih_c, Whh_c, bih_c, bhh_c,
            W_pm, b_pm, W_ps, b_ps,
            W_bm, b_bm, W_bs, b_bs,
            eps_phi + (size_t)t * NN * DD,
            eps_beta + (size_t)t * KK * DD);
        edge_kernel<<<NBLK, 128, 0, stream>>>(
            edges + (size_t)t * EE * 2,
            gumbel_u + (size_t)t * E2 * KK,
            phi, beta, wdec16, wpi16, b_pi, b_dec, partial);
    }
    finalize_kernel<<<1, 256, 0, stream>>>(partial, (float*)d_out);
}
